// BitNetFFN_9405978378867
// MI455X (gfx1250) — 8-bit matrix operand order repaired and checked by reading, compiled, not run on hardware
//
#include <hip/hip_runtime.h>
#include <hip/hip_bf16.h>
#include <math.h>


typedef int v8i __attribute__((ext_vector_type(8)));

#define EPSQ 1e-5f

__device__ __forceinline__ v8i load_frag(const signed char* __restrict__ P,
                                         int ktiles, int rtile, int kt, int lane) {
    const int4* p = (const int4*)(P + ((((size_t)rtile * (size_t)ktiles) + kt) << 10) + lane * 32);
    int4 lo = p[0];
    int4 hi = p[1];
    v8i v;
    v[0] = lo.x; v[1] = lo.y; v[2] = lo.z; v[3] = lo.w;
    v[4] = hi.x; v[5] = hi.y; v[6] = hi.z; v[7] = hi.w;
    return v;
}

__device__ __forceinline__ size_t a_shuf_idx(int row, int k, int K) {
    const int kk = k & 63;
    const int kt = k >> 6;
    const int chunk = kk >> 3;
    const int c = kk & 7;
    const int laneHalf = chunk & 1;
    const int chunkIdx = chunk >> 1;
    const int lane = laneHalf * 16 + (row & 15);
    return ((((size_t)(row >> 4)) * (size_t)(K >> 6) + kt) << 10) + lane * 32 + chunkIdx * 8 + c;
}

__device__ __forceinline__ size_t b_shuf_idx(int n, int k, int K) {
    const int kk = k & 63;
    const int kt = k >> 6;
    const int laneHalf = (kk >> 3) & 1;
    const int fragByte = (kk & 7) + ((kk >> 4) << 3);
    const int lane = laneHalf * 16 + (n & 15);
    return ((((size_t)(n >> 4)) * (size_t)(K >> 6) + kt) << 10) + lane * 32 + fragByte;
}


__global__ __launch_bounds__(256) void k_abs_partial(const float* __restrict__ w,
                                                     float* __restrict__ partial, int n) {
    __shared__ float red[256];
    float s = 0.f;
    for (size_t i = (size_t)blockIdx.x * 256 + threadIdx.x; i < (size_t)n;
         i += (size_t)gridDim.x * 256)
        s += fabsf(w[i]);
    red[threadIdx.x] = s;
    __syncthreads();
    for (int st = 128; st > 0; st >>= 1) {
        if ((int)threadIdx.x < st) red[threadIdx.x] += red[threadIdx.x + st];
        __syncthreads();
    }
    if (threadIdx.x == 0) partial[blockIdx.x] = red[0];
}

__global__ __launch_bounds__(256) void k_finalize_scale(const float* __restrict__ partial,
                                                        int nb, float inv_count,
                                                        float* __restrict__ out) {
    __shared__ float red[256];
    float s = 0.f;
    for (int i = threadIdx.x; i < nb; i += 256) s += partial[i];
    red[threadIdx.x] = s;
    __syncthreads();
    for (int st = 128; st > 0; st >>= 1) {
        if ((int)threadIdx.x < st) red[threadIdx.x] += red[threadIdx.x + st];
        __syncthreads();
    }
    if (threadIdx.x == 0) out[0] = red[0] * inv_count + EPSQ;
}

__global__ __launch_bounds__(256) void k_quant_weight_shufB(const float* __restrict__ w,
                                                            signed char* __restrict__ wq,
                                                            const float* __restrict__ scale,
                                                            int K, int kmask, int kshift, int n) {
    const float inv = 1.f / scale[0];
    for (size_t i = (size_t)blockIdx.x * 256 + threadIdx.x; i < (size_t)n;
         i += (size_t)gridDim.x * 256) {
        const int nrow = (int)(i >> kshift);
        const int k = (int)i & kmask;
        float v = rintf(w[i] * inv);
        v = fminf(fmaxf(v, -1.f), 1.f);
        wq[b_shuf_idx(nrow, k, K)] = (signed char)v;
    }
}


__device__ __forceinline__ float to_f32(float v) { return v; }
__device__ __forceinline__ float to_f32(__hip_bfloat16 v) { return __bfloat162float(v); }

template <typename T>
__global__ __launch_bounds__(256) void k_quant_rows_shufA(const T* __restrict__ X,
                                                          signed char* __restrict__ Xq,
                                                          float* __restrict__ rowscale, int K) {
    __shared__ float red[256];
    const int row = blockIdx.x;
    const T* xr = X + (size_t)row * (size_t)K;
    float m = 0.f;
    for (int i = threadIdx.x; i < K; i += 256) m = fmaxf(m, fabsf(to_f32(xr[i])));
    red[threadIdx.x] = m;
    __syncthreads();
    for (int st = 128; st > 0; st >>= 1) {
        if ((int)threadIdx.x < st)
            red[threadIdx.x] = fmaxf(red[threadIdx.x], red[threadIdx.x + st]);
        __syncthreads();
    }
    const float scale = fmaxf(red[0], EPSQ) * (1.f / 127.f);
    const float inv = 1.f / scale;
    if (threadIdx.x == 0) rowscale[row] = scale;
    for (int i = threadIdx.x; i < K; i += 256) {
        float v = rintf(to_f32(xr[i]) * inv);
        v = fminf(fmaxf(v, -128.f), 127.f);
        Xq[a_shuf_idx(row, i, K)] = (signed char)v;
    }
}


#define WMMA_IU8(A, B, C) \
    __builtin_amdgcn_wmma_i32_16x16x64_iu8(true, (A), true, (B), (C), false, false)

__global__ __launch_bounds__(256) void k_gemm_gateup(
    const signed char* __restrict__ Aq, const float* __restrict__ sA,
    const signed char* __restrict__ Bg, const signed char* __restrict__ Bu,
    const float* __restrict__ wsc,
    __hip_bfloat16* __restrict__ H, int N, int K) {
    const int lane = threadIdx.x & 31;
    const int wave = threadIdx.x >> 5;
    const int laneM = lane & 15;
    const int laneHalf = lane >> 4;
    const int rowBase = blockIdx.y * 64 + (wave & 1) * 32;
    const int colBase = blockIdx.x * 128 + (wave >> 1) * 32;
    const int mt = rowBase >> 4;
    const int nt = colBase >> 4;
    const int ktiles = K >> 6;

    v8i aG00 = {}, aG01 = {}, aG10 = {}, aG11 = {};
    v8i aU00 = {}, aU01 = {}, aU10 = {}, aU11 = {};

    for (int kt = 0; kt < ktiles; ++kt) {
        v8i a0 = load_frag(Aq, ktiles, mt, kt, lane);
        v8i a1 = load_frag(Aq, ktiles, mt + 1, kt, lane);
        v8i bg0 = load_frag(Bg, ktiles, nt, kt, lane);
        v8i bg1 = load_frag(Bg, ktiles, nt + 1, kt, lane);
        v8i bu0 = load_frag(Bu, ktiles, nt, kt, lane);
        v8i bu1 = load_frag(Bu, ktiles, nt + 1, kt, lane);
        aG00 = WMMA_IU8(a0, bg0, aG00);
        aG01 = WMMA_IU8(a0, bg1, aG01);
        aG10 = WMMA_IU8(a1, bg0, aG10);
        aG11 = WMMA_IU8(a1, bg1, aG11);
        aU00 = WMMA_IU8(a0, bu0, aU00);
        aU01 = WMMA_IU8(a0, bu1, aU01);
        aU10 = WMMA_IU8(a1, bu0, aU10);
        aU11 = WMMA_IU8(a1, bu1, aU11);
    }

    const float sg = wsc[0], su = wsc[1];
    v8i* accG[2][2] = {{&aG00, &aG01}, {&aG10, &aG11}};
    v8i* accU[2][2] = {{&aU00, &aU01}, {&aU10, &aU11}};
#pragma unroll
    for (int sm = 0; sm < 2; ++sm) {
#pragma unroll
        for (int sn = 0; sn < 2; ++sn) {
#pragma unroll
            for (int r = 0; r < 8; ++r) {
                const int row = rowBase + sm * 16 + r + laneHalf * 8;
                const int col = colBase + sn * 16 + laneM;
                const float ax = sA[row];
                const float g = (float)((*accG[sm][sn])[r]) * (ax * sg);
                const float u = (float)((*accU[sm][sn])[r]) * (ax * su);
                const float hval = u / (1.f + __expf(-g));
                H[(size_t)row * (size_t)N + col] = __float2bfloat16(hval);
            }
        }
    }
}


__global__ __launch_bounds__(256) void k_gemm_down(
    const signed char* __restrict__ Aq, const float* __restrict__ sA,
    const signed char* __restrict__ Bq, const float* __restrict__ wsc,
    float* __restrict__ Out, int N, int K) {
    const int lane = threadIdx.x & 31;
    const int wave = threadIdx.x >> 5;
    const int laneM = lane & 15;
    const int laneHalf = lane >> 4;
    const int rowBase = blockIdx.y * 64 + (wave & 1) * 32;
    const int colBase = blockIdx.x * 128 + (wave >> 1) * 32;
    const int mt = rowBase >> 4;
    const int nt = colBase >> 4;
    const int ktiles = K >> 6;

    v8i a00 = {}, a01 = {}, a10 = {}, a11 = {};

    for (int kt = 0; kt < ktiles; ++kt) {
        v8i fa0 = load_frag(Aq, ktiles, mt, kt, lane);
        v8i fa1 = load_frag(Aq, ktiles, mt + 1, kt, lane);
        v8i fb0 = load_frag(Bq, ktiles, nt, kt, lane);
        v8i fb1 = load_frag(Bq, ktiles, nt + 1, kt, lane);
        a00 = WMMA_IU8(fa0, fb0, a00);
        a01 = WMMA_IU8(fa0, fb1, a01);
        a10 = WMMA_IU8(fa1, fb0, a10);
        a11 = WMMA_IU8(fa1, fb1, a11);
    }

    const float sd = wsc[0];
    v8i* acc[2][2] = {{&a00, &a01}, {&a10, &a11}};
#pragma unroll
    for (int sm = 0; sm < 2; ++sm) {
#pragma unroll
        for (int sn = 0; sn < 2; ++sn) {
#pragma unroll
            for (int r = 0; r < 8; ++r) {
                const int row = rowBase + sm * 16 + r + laneHalf * 8;
                const int col = colBase + sn * 16 + laneM;
                Out[(size_t)row * (size_t)N + col] =
                    (float)((*acc[sm][sn])[r]) * (sA[row] * sd);
            }
        }
    }
}


extern "C" void kernel_launch(void* const* d_in, const int* in_sizes, int n_in,
                              void* d_out, int out_size, void* d_ws, size_t ws_size,
                              hipStream_t stream) {
    (void)in_sizes; (void)n_in; (void)out_size; (void)ws_size;

    const float* x  = (const float*)d_in[0];
    const float* wg = (const float*)d_in[1];
    const float* wu = (const float*)d_in[2];
    const float* wd = (const float*)d_in[3];
    float* out = (float*)d_out;

    const int M = 4 * 2048;
    const int D = 2048;
    const int F = 8192;
    const int WN = F * D;

    char* ws = (char*)d_ws;
    auto take = [&](size_t bytes) {
        char* p = ws;
        ws += (bytes + 1023) & ~(size_t)1023;
        return p;
    };
    signed char* xq  = (signed char*)take((size_t)M * D);
    signed char* wgq = (signed char*)take((size_t)WN);
    signed char* wuq = (signed char*)take((size_t)WN);
    signed char* wdq = (signed char*)take((size_t)WN);
    signed char* hq  = (signed char*)take((size_t)M * F);
    __hip_bfloat16* h = (__hip_bfloat16*)take((size_t)M * F * 2);
    float* sx      = (float*)take((size_t)M * 4);
    float* sh      = (float*)take((size_t)M * 4);
    float* wscale  = (float*)take(4 * 4);
    float* partial = (float*)take(1024 * 4);

    const float invWN = 1.f / (float)WN;

    k_abs_partial<<<1024, 256, 0, stream>>>(wg, partial, WN);
    k_finalize_scale<<<1, 256, 0, stream>>>(partial, 1024, invWN, wscale + 0);
    k_abs_partial<<<1024, 256, 0, stream>>>(wu, partial, WN);
    k_finalize_scale<<<1, 256, 0, stream>>>(partial, 1024, invWN, wscale + 1);
    k_abs_partial<<<1024, 256, 0, stream>>>(wd, partial, WN);
    k_finalize_scale<<<1, 256, 0, stream>>>(partial, 1024, invWN, wscale + 2);

    k_quant_weight_shufB<<<2048, 256, 0, stream>>>(wg, wgq, wscale + 0, D, D - 1, 11, WN);
    k_quant_weight_shufB<<<2048, 256, 0, stream>>>(wu, wuq, wscale + 1, D, D - 1, 11, WN);
    k_quant_weight_shufB<<<2048, 256, 0, stream>>>(wd, wdq, wscale + 2, F, F - 1, 13, WN);

    k_quant_rows_shufA<float><<<M, 256, 0, stream>>>(x, xq, sx, D);

    {
        dim3 grid(F / 128, M / 64);
        k_gemm_gateup<<<grid, 256, 0, stream>>>(xq, sx, wgq, wuq, wscale, h, F, D);
    }

    k_quant_rows_shufA<__hip_bfloat16><<<M, 256, 0, stream>>>(h, hq, sh, F);

    {
        dim3 grid(D / 128, M / 64);
        k_gemm_down<<<grid, 256, 0, stream>>>(hq, sh, wdq, wscale + 2, out, D, F);
    }
}
